// InterpretableMultiHeadAttention_16518444221133
// MI455X (gfx1250) — matrix operand order repaired and checked by reading, not compiled, not run on hardware
//
#include <hip/hip_runtime.h>
#include <stdint.h>
#include <stddef.h>

// ---------------------------------------------------------------------------
// MHA forward for MI455X (gfx1250, wave32, WMMA bf16 16x16x32, f32 accum).
// d_out = [ out : B*L*D f32 ][ attn : B*H*L*L f32 ]
// ---------------------------------------------------------------------------

#define BATCH    2
#define SEQ      4096
#define EMBED    512
#define HEADS    8
#define HEAD_DIM 64

typedef __bf16 bf16;
typedef __attribute__((ext_vector_type(16))) __bf16 v16bf;
typedef __attribute__((ext_vector_type(8)))  float  v8f;

__device__ __forceinline__ bf16 f2bf(float f) {
    return static_cast<bf16>(f);          // RNE; lowers to native cvt on gfx1250
}

__device__ __forceinline__ v8f wmma_bf16(v16bf a, v16bf b, v8f c) {
    return __builtin_amdgcn_wmma_f32_16x16x32_bf16(
        /*neg_a=*/false, a, /*neg_b=*/false, b,
        /*c_mod=*/(short)0, c, /*reuse_a=*/false, /*reuse_b=*/false);
}

// ---------------------------------------------------------------------------
// Kernel 1: projection  y = scale * (X @ W^T + bias), stored bf16.
//   vtrans==0 : dst[b,h,l,hd]  (head-major, rows = sequence)   for Q, K
//   vtrans==1 : dst[b,h,hd,l]  (head-major, rows = head dim)   for V
// X: f32 [B*L, EMBED] row-major, W: f32 [EMBED, EMBED] row-major (nn.Linear).
// One 16x16 output tile per wave; K-loop over EMBED in steps of 32.
// ---------------------------------------------------------------------------
__global__ void __launch_bounds__(256)
proj_kernel(const float* __restrict__ X, const float* __restrict__ W,
            const float* __restrict__ bias, bf16* __restrict__ dst,
            float scale, int vtrans)
{
    const int w    = threadIdx.x >> 5;
    const int lane = threadIdx.x & 31;
    const int half = lane >> 4;
    const int l16  = lane & 15;
    const int m0 = blockIdx.y * 32 + (w >> 2) * 16;   // rows in [B*L]
    const int n0 = blockIdx.x * 64 + (w & 3) * 16;    // cols in [EMBED]

    const float* xa_base = X + (size_t)(m0 + l16) * EMBED;
    const float* wb_base = W + (size_t)(n0 + l16) * EMBED;

    v8f c = {};
    for (int k0 = 0; k0 < EMBED; k0 += 32) {
        if (k0 + 32 < EMBED) __builtin_prefetch(xa_base + k0 + 32, 0, 0);
        v16bf a, b;
        const float* xa = xa_base + k0 + half * 8;
        #pragma unroll
        for (int j = 0; j < 8; ++j) { a[j] = f2bf(xa[j]); a[j + 8] = f2bf(xa[16 + j]); }
        const float* wb = wb_base + k0 + half * 8;
        #pragma unroll
        for (int j = 0; j < 8; ++j) { b[j] = f2bf(wb[j]); b[j + 8] = f2bf(wb[16 + j]); }
        c = wmma_bf16(a, b, c);
    }

    const float bn = bias[n0 + l16];
    const int h  = (n0 + l16) >> 6;       // head
    const int hd = (n0 + l16) & 63;       // dim within head
    #pragma unroll
    for (int r = 0; r < 8; ++r) {
        const int m  = m0 + r + half * 8;        // row in [B*L]
        const int bi = m >> 12;                  // / SEQ
        const int l  = m & (SEQ - 1);
        const float v = (c[r] + bn) * scale;
        const size_t bh = (size_t)(bi * HEADS + h);
        if (vtrans)
            dst[(bh * HEAD_DIM + hd) * SEQ + l] = f2bf(v);
        else
            dst[(bh * SEQ + l) * HEAD_DIM + hd] = f2bf(v);
    }
}

// ---------------------------------------------------------------------------
// Kernel 2: raw scores  S[bh,q,k] = sum_d Qb[bh,q,d] * Kb[bh,k,d]
// (0.125 scale already folded into Qb). Tiles fully above the causal
// diagonal are skipped (wave-uniform branch -> EXEC stays all-ones).
// ---------------------------------------------------------------------------
__global__ void __launch_bounds__(256)
scores_kernel(const bf16* __restrict__ Qb, const bf16* __restrict__ Kb,
              float* __restrict__ S)
{
    const int w    = threadIdx.x >> 5;
    const int lane = threadIdx.x & 31;
    const int half = lane >> 4;
    const int l16  = lane & 15;
    const int bh = blockIdx.z;
    const int q0 = blockIdx.y * 32 + (w >> 2) * 16;
    const int k0 = blockIdx.x * 64 + (w & 3) * 16;
    if (k0 > q0 + 15) return;                         // fully masked tile

    const bf16* Q = Qb + (size_t)bh * SEQ * HEAD_DIM;
    const bf16* K = Kb + (size_t)bh * SEQ * HEAD_DIM;

    v8f c = {};
    #pragma unroll
    for (int kd = 0; kd < HEAD_DIM; kd += 32) {
        v16bf a, b;
        const bf16* qa = Q + (size_t)(q0 + l16) * HEAD_DIM + kd + half * 8;
        #pragma unroll
        for (int j = 0; j < 8; ++j) { a[j] = qa[j]; a[j + 8] = qa[16 + j]; }
        const bf16* kb = K + (size_t)(k0 + l16) * HEAD_DIM + kd + half * 8;
        #pragma unroll
        for (int j = 0; j < 8; ++j) { b[j] = kb[j]; b[j + 8] = kb[16 + j]; }
        c = wmma_bf16(a, b, c);
    }

    float* Sp = S + ((size_t)bh * SEQ + q0) * SEQ + k0;
    #pragma unroll
    for (int r = 0; r < 8; ++r)
        Sp[(size_t)(r + half * 8) * SEQ + l16] = c[r];
}

// ---------------------------------------------------------------------------
// Kernel 3: in-place causal softmax over each row of S. One 256-thread
// block per (bh, q) row; reads only k <= q, writes 0 for k > q.
// ---------------------------------------------------------------------------
__global__ void __launch_bounds__(256)
softmax_kernel(float* __restrict__ S)
{
    __shared__ float red[256];
    const int row = blockIdx.x;
    const int q   = row & (SEQ - 1);
    float* p = S + (size_t)row * SEQ;
    const int n   = q + 1;
    const int tid = threadIdx.x;

    float m = -3.402823466e38f;
    for (int k = tid; k < n; k += 256) m = fmaxf(m, p[k]);
    red[tid] = m; __syncthreads();
    #pragma unroll
    for (int s = 128; s > 0; s >>= 1) {
        if (tid < s) red[tid] = fmaxf(red[tid], red[tid + s]);
        __syncthreads();
    }
    m = red[0]; __syncthreads();

    float sum = 0.0f;
    for (int k = tid; k < n; k += 256) sum += __expf(p[k] - m);
    red[tid] = sum; __syncthreads();
    #pragma unroll
    for (int s = 128; s > 0; s >>= 1) {
        if (tid < s) red[tid] += red[tid + s];
        __syncthreads();
    }
    const float inv = 1.0f / red[0];

    for (int k = tid; k < SEQ; k += 256)
        p[k] = (k < n) ? __expf(p[k] - m) * inv : 0.0f;
}

// ---------------------------------------------------------------------------
// Kernel 4: O = attn @ V.
//   attn (f32) tiles are staged block-wide into LDS with
//   GLOBAL_LOAD_ASYNC_TO_LDS_B128 (ASYNCcnt) so the 4 waves sharing a q-tile
//   read the dominant 1 GiB stream once per block instead of 4x.
//   V is bf16 transposed head-major: Vt[bh][hd][l] -> B fragments are
//   two contiguous 16-byte runs per lane.
//   K-loop trip count is block-uniform (attn is exactly 0 for k>q, so the
//   extra columns of partial diagonal tiles contribute nothing).
// ---------------------------------------------------------------------------
#define PT_STRIDE 36   // 32 floats + 4 pad (144 B rows: 16B-aligned, bank spread)

__global__ void __launch_bounds__(256)
attnv_kernel(const float* __restrict__ P, const bf16* __restrict__ Vt,
             bf16* __restrict__ Ob)
{
    __shared__ float Ptile[32 * PT_STRIDE];

    const int tid  = threadIdx.x;
    const int w    = tid >> 5;
    const int lane = tid & 31;
    const int half = lane >> 4;
    const int l16  = lane & 15;
    const int bh = blockIdx.y;
    const int qB = blockIdx.x * 32;            // block's first q row
    const int n0 = (w & 3) * 16;

    // per-thread async-copy slot: 16 B = 4 floats of one attn row
    const int cRow = tid >> 3;                 // 0..31
    const int cCol = (tid & 7) * 4;            // 0,4,...,28
    const float* gRow = P + ((size_t)bh * SEQ + qB + cRow) * SEQ + cCol;
    const uint32_t ldsOff =
        (uint32_t)(uintptr_t)(&Ptile[0]) + (uint32_t)(cRow * PT_STRIDE + cCol) * 4u;

    const int aRow = (w >> 2) * 16 + l16;      // this wave's A-fragment row
    const float* pa = &Ptile[aRow * PT_STRIDE];
    const bf16* vb_base = Vt + ((size_t)bh * HEAD_DIM + n0 + l16) * SEQ;

    v8f c = {};
    const int kiters = blockIdx.x + 1;         // k-tiles of 32 covering k <= qB+31
    for (int it = 0; it < kiters; ++it) {
        const int k0 = it * 32;
        // stage 32x32 f32 attn tile: 256 threads x 16 B, ASYNCcnt-tracked
        asm volatile("global_load_async_to_lds_b128 %0, %1, off"
                     :: "v"(ldsOff), "v"(gRow + k0) : "memory");
        asm volatile("s_wait_asynccnt 0x0" ::: "memory");
        __syncthreads();

        v16bf a, b;
        #pragma unroll
        for (int j = 0; j < 8; ++j) {
            a[j]     = f2bf(pa[half * 8 + j]);
            a[j + 8] = f2bf(pa[16 + half * 8 + j]);
        }
        const bf16* vb = vb_base + k0 + half * 8;
        #pragma unroll
        for (int j = 0; j < 8; ++j) { b[j] = vb[j]; b[j + 8] = vb[16 + j]; }
        c = wmma_bf16(a, b, c);
        __syncthreads();                       // LDS reused next iteration
    }

    const int bi = bh >> 3;                    // / HEADS
    const int h  = bh & 7;
    #pragma unroll
    for (int r = 0; r < 8; ++r) {
        const int l = qB + (w >> 2) * 16 + r + half * 8;
        Ob[((size_t)(bi * SEQ + l)) * EMBED + h * HEAD_DIM + n0 + l16] = f2bf(c[r]);
    }
}

// ---------------------------------------------------------------------------
// Kernel 5: final projection  out = Ob @ Wo^T + bo   (bf16 X, f32 W/out)
// ---------------------------------------------------------------------------
__global__ void __launch_bounds__(256)
outproj_kernel(const bf16* __restrict__ X, const float* __restrict__ W,
               const float* __restrict__ bias, float* __restrict__ out)
{
    const int w    = threadIdx.x >> 5;
    const int lane = threadIdx.x & 31;
    const int half = lane >> 4;
    const int l16  = lane & 15;
    const int m0 = blockIdx.y * 32 + (w >> 2) * 16;
    const int n0 = blockIdx.x * 64 + (w & 3) * 16;

    const bf16*  xa_base = X + (size_t)(m0 + l16) * EMBED;
    const float* wb_base = W + (size_t)(n0 + l16) * EMBED;

    v8f c = {};
    for (int k0 = 0; k0 < EMBED; k0 += 32) {
        if (k0 + 32 < EMBED) __builtin_prefetch(xa_base + k0 + 32, 0, 0);
        v16bf a, b;
        const bf16* xa = xa_base + k0 + half * 8;
        #pragma unroll
        for (int j = 0; j < 8; ++j) { a[j] = xa[j]; a[j + 8] = xa[16 + j]; }
        const float* wb = wb_base + k0 + half * 8;
        #pragma unroll
        for (int j = 0; j < 8; ++j) { b[j] = f2bf(wb[j]); b[j + 8] = f2bf(wb[16 + j]); }
        c = wmma_bf16(a, b, c);
    }

    const float bn = bias[n0 + l16];
    #pragma unroll
    for (int r = 0; r < 8; ++r) {
        const int m = m0 + r + half * 8;
        out[(size_t)m * EMBED + n0 + l16] = c[r] + bn;
    }
}

// ---------------------------------------------------------------------------
extern "C" void kernel_launch(void* const* d_in, const int* in_sizes, int n_in,
                              void* d_out, int out_size, void* d_ws, size_t ws_size,
                              hipStream_t stream)
{
    const float* query = (const float*)d_in[0];
    const float* key   = (const float*)d_in[1];
    const float* value = (const float*)d_in[2];
    const float* Wq = (const float*)d_in[3];
    const float* bq = (const float*)d_in[4];
    const float* Wk = (const float*)d_in[5];
    const float* bk = (const float*)d_in[6];
    const float* Wv = (const float*)d_in[7];
    const float* bv = (const float*)d_in[8];
    const float* Wo = (const float*)d_in[9];
    const float* bo = (const float*)d_in[10];

    float* out0 = (float*)d_out;                                   // [B,L,D]
    float* attn = out0 + (size_t)BATCH * SEQ * EMBED;              // [B,H,L,L]

    const size_t QKV = (size_t)BATCH * SEQ * EMBED;                // per-tensor elems
    bf16* Qb = (bf16*)d_ws;                                        // [bh][l][hd]
    bf16* Kb = Qb + QKV;                                           // [bh][l][hd]
    bf16* Vt = Kb + QKV;                                           // [bh][hd][l]
    bf16* Ob = Vt + QKV;                                           // [b][l][D]

    const dim3 blk(256);
    const dim3 gproj(EMBED / 64, (BATCH * SEQ) / 32);              // (8, 256)

    // scale 1/sqrt(Hd) = 0.125 folded exactly (power of two) into Q.
    proj_kernel<<<gproj, blk, 0, stream>>>(query, Wq, bq, Qb, 0.125f, 0);
    proj_kernel<<<gproj, blk, 0, stream>>>(key,   Wk, bk, Kb, 1.0f,   0);
    proj_kernel<<<gproj, blk, 0, stream>>>(value, Wv, bv, Vt, 1.0f,   1);

    const dim3 gsc(SEQ / 64, SEQ / 32, BATCH * HEADS);             // (64,128,16)
    scores_kernel<<<gsc, blk, 0, stream>>>(Qb, Kb, attn);

    softmax_kernel<<<dim3(BATCH * HEADS * SEQ), blk, 0, stream>>>(attn);

    const dim3 gav(SEQ / 32, BATCH * HEADS);                       // (128, 16)
    attnv_kernel<<<gav, blk, 0, stream>>>(attn, Vt, Ob);

    outproj_kernel<<<gproj, blk, 0, stream>>>(Ob, Wo, bo, out0);
}
